// CrossAttentionWithPrior_41403484733747
// MI455X (gfx1250) — matrix operand order repaired and checked by reading, not compiled, not run on hardware
//
#include <hip/hip_runtime.h>

#define SQ   2048
#define SKV  2048
#define DIMN 1024
#define NH   16
#define HD   64
#define NB   2
#define EPSC 1e-8f
#define SCALEC 0.125f

typedef __bf16 bf16_t;
typedef __attribute__((ext_vector_type(16))) __bf16 v16bf;
typedef __attribute__((ext_vector_type(8)))  float  v8f;
typedef __attribute__((ext_vector_type(4)))  float  v4f;

__device__ __forceinline__ v8f wmma_bf16(v16bf a, v16bf b, v8f c) {
  return __builtin_amdgcn_wmma_f32_16x16x32_bf16(false, a, false, b, (short)0, c,
                                                 false, false);
}

__device__ __forceinline__ v8f zero8() {
  v8f c;
#pragma unroll
  for (int i = 0; i < 8; ++i) c[i] = 0.0f;
  return c;
}

// ---- fragment loaders --------------------------------------------------
// A (16x32, bf16): lane l -> row m=l&15; K elems {kb..kb+7, kb+16..kb+23},
// kb = (l<16)?0:8.  B (32x16): lane l -> col n=l&15; K elems {kb..kb+7, kb+16..kb+23},
// kb = (l<16)?0:8.  (CDNA5 ISA 7.12.2)

__device__ __forceinline__ v16bf frag_a_f32(const float* __restrict__ base,
                                            int ld, int m0, int k0, int lane) {
  const float* p =
      base + (size_t)(m0 + (lane & 15)) * ld + k0 + ((lane & 16) ? 8 : 0);
  v4f x0 = *(const v4f*)(p);
  v4f x1 = *(const v4f*)(p + 4);
  v4f x2 = *(const v4f*)(p + 16);
  v4f x3 = *(const v4f*)(p + 20);
  v16bf a;
#pragma unroll
  for (int i = 0; i < 4; ++i) {
    a[i]      = (bf16_t)x0[i];
    a[4 + i]  = (bf16_t)x1[i];
    a[8 + i]  = (bf16_t)x2[i];
    a[12 + i] = (bf16_t)x3[i];
  }
  return a;
}

__device__ __forceinline__ v16bf frag_b_f32(const float* __restrict__ base,
                                            int ld, int n0, int k0, int lane) {
  const float* p =
      base + (size_t)(n0 + (lane & 15)) * ld + k0 + ((lane & 16) ? 8 : 0);
  v4f x0 = *(const v4f*)(p);
  v4f x1 = *(const v4f*)(p + 4);
  v4f x2 = *(const v4f*)(p + 16);
  v4f x3 = *(const v4f*)(p + 20);
  v16bf b;
#pragma unroll
  for (int i = 0; i < 4; ++i) {
    b[i]      = (bf16_t)x0[i];
    b[4 + i]  = (bf16_t)x1[i];
    b[8 + i]  = (bf16_t)x2[i];
    b[12 + i] = (bf16_t)x3[i];
  }
  return b;
}

__device__ __forceinline__ v16bf frag_a_bf16(const bf16_t* __restrict__ base,
                                             int ld, int m0, int k0, int lane) {
  const bf16_t* p =
      base + (size_t)(m0 + (lane & 15)) * ld + k0 + ((lane & 16) ? 8 : 0);
  union { v16bf v; uint4 u[2]; } t;
  t.u[0] = *(const uint4*)(p);        // K kb..kb+7
  t.u[1] = *(const uint4*)(p + 16);   // K kb+16..kb+23
  return t.v;
}

__device__ __forceinline__ v16bf frag_b_bf16(const bf16_t* __restrict__ base,
                                             int ld, int n0, int k0, int lane) {
  const bf16_t* p =
      base + (size_t)(n0 + (lane & 15)) * ld + k0 + ((lane & 16) ? 8 : 0);
  union { v16bf v; uint4 u[2]; } t;
  t.u[0] = *(const uint4*)(p);        // K kb..kb+7
  t.u[1] = *(const uint4*)(p + 16);   // K kb+16..kb+23
  return t.v;
}

// 16-lane-group butterfly reductions (masks < 16 stay within half-wave)
__device__ __forceinline__ float rmax16(float v) {
#pragma unroll
  for (int m = 1; m < 16; m <<= 1) v = fmaxf(v, __shfl_xor(v, m, 32));
  return v;
}
__device__ __forceinline__ float rsum16(float v) {
#pragma unroll
  for (int m = 1; m < 16; m <<= 1) v += __shfl_xor(v, m, 32);
  return v;
}

// ---- kernel 1: q projection + RoPE ------------------------------------
__global__ __launch_bounds__(256) void k_qproj_rope(
    const float* __restrict__ qx, const float* __restrict__ Wq,
    const float* __restrict__ qfreqs, bf16_t* __restrict__ q16) {
  const int lane = threadIdx.x & 31;
  const int wave = threadIdx.x >> 5;
  const int m0 = blockIdx.x * 128 + wave * 16;
  const int n0 = blockIdx.y * 64;

  v8f acc[4];
#pragma unroll
  for (int t = 0; t < 4; ++t) acc[t] = zero8();

  for (int k0 = 0; k0 < DIMN; k0 += 32) {
    v16bf a = frag_a_f32(qx, DIMN, m0, k0, lane);
#pragma unroll
    for (int t = 0; t < 4; ++t) {
      v16bf b = frag_b_f32(Wq, DIMN, n0 + t * 16, k0, lane);
      acc[t] = wmma_bf16(a, b, acc[t]);
    }
  }

  const int rowoff = (lane & 16) ? 8 : 0;
  const int col = lane & 15;
#pragma unroll
  for (int t = 0; t < 4; ++t) {
#pragma unroll
    for (int r = 0; r < 8; ++r) {
      int m = m0 + r + rowoff;
      int n = n0 + t * 16 + col;
      float v = acc[t][r];
      float pv = __shfl_xor(v, 1, 32);  // partner column (hd pair)
      int b = m >> 11, sq = m & (SQ - 1);
      int h = n >> 6, hd = n & 63;
      const float* f = qfreqs + ((size_t)b * SQ + sq) * HD;
      float cs = f[hd >> 1], sn = f[32 + (hd >> 1)];
      float o = (n & 1) ? (pv * sn + v * cs) : (v * cs - pv * sn);
      q16[(((size_t)(b * NH + h)) * SQ + sq) * HD + hd] = (bf16_t)o;
    }
  }
}

// ---- kernel 2: kv projection, RoPE on K, V transposed ------------------
__global__ __launch_bounds__(256) void k_kvproj_rope(
    const float* __restrict__ kvx, const float* __restrict__ Wkv,
    const float* __restrict__ kfreqs, bf16_t* __restrict__ k16,
    bf16_t* __restrict__ v16T) {
  const int lane = threadIdx.x & 31;
  const int wave = threadIdx.x >> 5;
  const int m0 = blockIdx.x * 128 + wave * 16;
  const int n0 = blockIdx.y * 64;

  v8f acc[4];
#pragma unroll
  for (int t = 0; t < 4; ++t) acc[t] = zero8();

  for (int k0 = 0; k0 < DIMN; k0 += 32) {
    v16bf a = frag_a_f32(kvx, DIMN, m0, k0, lane);
#pragma unroll
    for (int t = 0; t < 4; ++t) {
      v16bf b = frag_b_f32(Wkv, DIMN, n0 + t * 16, k0, lane);
      acc[t] = wmma_bf16(a, b, acc[t]);
    }
  }

  const int rowoff = (lane & 16) ? 8 : 0;
  const int col = lane & 15;
#pragma unroll
  for (int t = 0; t < 4; ++t) {
    int c2 = (n0 + t * 16) >> 10;  // 0 = K, 1 = V (uniform per tile)
#pragma unroll
    for (int r = 0; r < 8; ++r) {
      int m = m0 + r + rowoff;
      int n = n0 + t * 16 + col;
      float v = acc[t][r];
      int b = m >> 11, skv = m & (SKV - 1);
      int rem = n & 1023;
      int h = rem >> 6, hd = rem & 63;
      if (c2 == 0) {
        float pv = __shfl_xor(v, 1, 32);
        const float* f = kfreqs + ((size_t)b * SKV + skv) * HD;
        float cs = f[hd >> 1], sn = f[32 + (hd >> 1)];
        float o = (n & 1) ? (pv * sn + v * cs) : (v * cs - pv * sn);
        k16[(((size_t)(b * NH + h)) * SKV + skv) * HD + hd] = (bf16_t)o;
      } else {
        v16T[(((size_t)(b * NH + h)) * HD + hd) * SKV + skv] = (bf16_t)v;
      }
    }
  }
}

// ---- kernel 3: attention (3 kv sweeps, write attn NT, PV via WMMA) -----
__global__ __launch_bounds__(256) void k_attn(
    const bf16_t* __restrict__ q16, const bf16_t* __restrict__ k16,
    const bf16_t* __restrict__ v16T, const float* __restrict__ mask,
    const float* __restrict__ prior, float* __restrict__ attn_out,
    bf16_t* __restrict__ o16) {
  __shared__ float ust[8][16][32];  // per-wave attn tile staging (16KB)

  const int lane = threadIdx.x & 31;
  const int wave = threadIdx.x >> 5;
  const int qb = blockIdx.x & 15;
  const int h = (blockIdx.x >> 4) & 15;
  const int b = blockIdx.x >> 8;
  const int q0 = qb * 128 + wave * 16;

  const bf16_t* qbh = q16 + (size_t)(b * NH + h) * SQ * HD;
  const bf16_t* kbh = k16 + (size_t)(b * NH + h) * SKV * HD;
  const bf16_t* vbh = v16T + (size_t)(b * NH + h) * HD * SKV;
  const float* maskb = mask + (size_t)b * SQ * SKV;
  const float* priorb = prior + (size_t)b * SQ * SKV;
  float* attnbh = attn_out + (size_t)(b * NH + h) * SQ * SKV;

  const v16bf aq0 = frag_a_bf16(qbh, HD, q0, 0, lane);
  const v16bf aq1 = frag_a_bf16(qbh, HD, q0, 32, lane);

  const int rowoff = (lane & 16) ? 8 : 0;
  const int col = lane & 15;

  auto score = [&](int kvb, float* s) {
    v16bf b0 = frag_b_bf16(kbh, HD, kvb, 0, lane);
    v16bf b1 = frag_b_bf16(kbh, HD, kvb, 32, lane);
    v8f c = zero8();
    c = wmma_bf16(aq0, b0, c);
    c = wmma_bf16(aq1, b1, c);
#pragma unroll
    for (int r = 0; r < 8; ++r) {
      int q = q0 + r + rowoff, kv = kvb + col;
      s[r] = c[r] * SCALEC + maskb[(size_t)q * SKV + kv];
    }
  };

  // -- pass 1: online row max / sum-exp
  float mx[8], sm[8];
#pragma unroll
  for (int r = 0; r < 8; ++r) { mx[r] = -3.0e38f; sm[r] = 0.0f; }
  for (int j = 0; j < SKV / 16; ++j) {
    float s[8];
    score(j * 16, s);
#pragma unroll
    for (int r = 0; r < 8; ++r) {
      float nm = fmaxf(mx[r], rmax16(s[r]));
      float ts = rsum16(__expf(s[r] - nm));
      sm[r] = sm[r] * __expf(mx[r] - nm) + ts;
      mx[r] = nm;
    }
  }
  float rsn[8];
#pragma unroll
  for (int r = 0; r < 8; ++r) rsn[r] = 1.0f / sm[r];

  // -- pass 2: row-sum of u = clip(p,eps)*clip(prior,eps)
  float Ua[8];
#pragma unroll
  for (int r = 0; r < 8; ++r) Ua[r] = 0.0f;
  for (int j = 0; j < SKV / 16; ++j) {
    float s[8];
    score(j * 16, s);
#pragma unroll
    for (int r = 0; r < 8; ++r) {
      int q = q0 + r + rowoff, kv = j * 16 + col;
      float p = __expf(s[r] - mx[r]) * rsn[r];
      Ua[r] += fmaxf(p, EPSC) * fmaxf(priorb[(size_t)q * SKV + kv], EPSC);
    }
  }
  float invU[8];
#pragma unroll
  for (int r = 0; r < 8; ++r) invU[r] = 1.0f / rsum16(Ua[r]);

  // -- pass 3: write attn (non-temporal) + accumulate attn @ V -----------
  v8f acco[4];
#pragma unroll
  for (int t = 0; t < 4; ++t) acco[t] = zero8();

  for (int j2 = 0; j2 < SKV / 32; ++j2) {
    int kvb = j2 * 32;
#pragma unroll
    for (int half = 0; half < 2; ++half) {
      float s[8];
      score(kvb + half * 16, s);
#pragma unroll
      for (int r = 0; r < 8; ++r) {
        int q = q0 + r + rowoff, kv = kvb + half * 16 + col;
        float p = __expf(s[r] - mx[r]) * rsn[r];
        float u = fmaxf(p, EPSC) * fmaxf(priorb[(size_t)q * SKV + kv], EPSC);
        float at = u * invU[r];
        // 537MB stream, never re-read: bypass cache residency (TH=NT)
        __builtin_nontemporal_store(at, &attnbh[(size_t)q * SKV + kv]);
        ust[wave][r + rowoff][half * 16 + col] = at;
      }
    }
    // relayout to A-fragment (DS ops from one wave are in-order)
    v16bf au;
    {
      int row = lane & 15;
      int kb = (lane & 16) ? 8 : 0;
#pragma unroll
      for (int i = 0; i < 8; ++i) {
        au[i] = (bf16_t)ust[wave][row][kb + i];
        au[8 + i] = (bf16_t)ust[wave][row][kb + 16 + i];
      }
    }
#pragma unroll
    for (int t = 0; t < 4; ++t) {
      v16bf bv = frag_b_bf16(vbh, SKV, t * 16, kvb, lane);
      acco[t] = wmma_bf16(au, bv, acco[t]);
    }
  }

#pragma unroll
  for (int t = 0; t < 4; ++t) {
#pragma unroll
    for (int r = 0; r < 8; ++r) {
      int q = q0 + r + rowoff;
      int hd = t * 16 + col;
      o16[((size_t)b * SQ + q) * DIMN + h * HD + hd] = (bf16_t)acco[t][r];
    }
  }
}

// ---- kernel 4: output projection + bias --------------------------------
__global__ __launch_bounds__(256) void k_outproj(
    const bf16_t* __restrict__ o16, const float* __restrict__ Wproj,
    const float* __restrict__ bproj, float* __restrict__ out_x) {
  const int lane = threadIdx.x & 31;
  const int wave = threadIdx.x >> 5;
  const int m0 = blockIdx.x * 128 + wave * 16;
  const int n0 = blockIdx.y * 64;

  v8f acc[4];
#pragma unroll
  for (int t = 0; t < 4; ++t) acc[t] = zero8();

  for (int k0 = 0; k0 < DIMN; k0 += 32) {
    v16bf a = frag_a_bf16(o16, DIMN, m0, k0, lane);
#pragma unroll
    for (int t = 0; t < 4; ++t) {
      v16bf b = frag_b_f32(Wproj, DIMN, n0 + t * 16, k0, lane);
      acc[t] = wmma_bf16(a, b, acc[t]);
    }
  }

  const int rowoff = (lane & 16) ? 8 : 0;
  const int col = lane & 15;
#pragma unroll
  for (int t = 0; t < 4; ++t) {
#pragma unroll
    for (int r = 0; r < 8; ++r) {
      int m = m0 + r + rowoff;
      int n = n0 + t * 16 + col;
      float xo = acc[t][r] + bproj[n];
      __builtin_nontemporal_store(xo, &out_x[(size_t)m * DIMN + n]);
    }
  }
}

// ---- host launch -------------------------------------------------------
extern "C" void kernel_launch(void* const* d_in, const int* in_sizes, int n_in,
                              void* d_out, int out_size, void* d_ws,
                              size_t ws_size, hipStream_t stream) {
  (void)in_sizes; (void)n_in; (void)out_size; (void)ws_size;

  const float* q_x     = (const float*)d_in[0];
  const float* kv_x    = (const float*)d_in[1];
  const float* q_freqs = (const float*)d_in[2];
  const float* k_freqs = (const float*)d_in[3];
  const float* mask    = (const float*)d_in[4];
  const float* prior   = (const float*)d_in[5];
  const float* Wq      = (const float*)d_in[6];
  const float* Wkv     = (const float*)d_in[7];
  const float* Wproj   = (const float*)d_in[8];
  const float* bproj   = (const float*)d_in[9];

  float* out_x    = (float*)d_out;
  float* out_attn = out_x + (size_t)NB * SQ * DIMN;

  const size_t nelem = (size_t)NB * NH * SQ * HD;  // 4,194,304
  bf16_t* q16  = (bf16_t*)d_ws;
  bf16_t* k16  = q16 + nelem;
  bf16_t* v16T = k16 + nelem;
  bf16_t* o16  = v16T + nelem;

  dim3 blk(256);
  k_qproj_rope<<<dim3(NB * SQ / 128, DIMN / 64), blk, 0, stream>>>(
      q_x, Wq, q_freqs, q16);
  k_kvproj_rope<<<dim3(NB * SKV / 128, 2 * DIMN / 64), blk, 0, stream>>>(
      kv_x, Wkv, k_freqs, k16, v16T);
  k_attn<<<dim3(NB * NH * (SQ / 128)), blk, 0, stream>>>(
      q16, k16, v16T, mask, prior, out_attn, o16);
  k_outproj<<<dim3(NB * SQ / 128, DIMN / 64), blk, 0, stream>>>(
      o16, Wproj, bproj, out_x);
}
